// Block_35880156790920
// MI455X (gfx1250) — matrix operand order repaired and checked by reading, not compiled, not run on hardware
//
#include <hip/hip_runtime.h>
#include <hip/hip_bf16.h>

// ---------------------------------------------------------------------------
// Transformer block fwd (cosine attention) for gfx1250 (MI455X), bf16 WMMA,
// TDM (tensor_load_to_lds) double-buffered GEMM pipeline.
// ---------------------------------------------------------------------------

#define B_    2
#define S_    2048
#define D_    1024
#define H_    16
#define HD_   64
#define MLP_  4096
#define TOK   (B_ * S_)          // 4096 rows
#define LOG_MAX_ 4.6051701860f   // log(1/0.01)

typedef __bf16 bf16;
typedef __attribute__((ext_vector_type(16))) __bf16 bf16x16;
typedef __attribute__((ext_vector_type(8)))  float  f32x8;
typedef unsigned int v4u __attribute__((ext_vector_type(4)));
typedef unsigned int v8u __attribute__((ext_vector_type(8)));

union FragU { uint4 u[2]; bf16x16 v; };

static __device__ __forceinline__ f32x8 wmma_bf16(const FragU& a, const FragU& b, f32x8 c) {
    return __builtin_amdgcn_wmma_f32_16x16x32_bf16(false, a.v, false, b.v, (short)0, c, false, false);
}

// ---------------------------------------------------------------------------
// TDM: async DMA of a 2-D bf16 tile (rows x 32 elems, row stride = K elems)
// from global into LDS, padding each 32-elem row to a 40-elem LDS stride
// (pad_interval: 16 DWORDs, pad_amount: 4 DWORDs). Tracked on TENSORcnt.
// D# built per CDNA5 ISA 8.3/8.4; 2-D tensor -> two SGPR groups.
// ---------------------------------------------------------------------------
static __device__ __forceinline__ void tdm_load_tile32(uint32_t lds_off, uint64_t gaddr,
                                                       uint32_t rows, uint32_t stride_elems) {
    const uint32_t td0 = 32u;          // tensor_dim0 == tile_dim0 (no OOB)
    const uint32_t td1 = rows;         // tensor_dim1 == tile_dim1
    v4u g0;
    g0[0] = 1u;                                        // count=1 (valid user D#)
    g0[1] = lds_off;                                   // lds_addr (bytes)
    g0[2] = (uint32_t)gaddr;                           // global_addr[31:0]
    g0[3] = (uint32_t)(gaddr >> 32) | (2u << 30);      // global_addr[56:32] | type=2
    v8u g1;
    g1[0] = (1u << 16)      // data_size = 1 (2 bytes)
          | (1u << 20)      // pad_enable
          | (3u << 22)      // pad_interval = 16 DWORDs (one 32-elem bf16 row)
          | (3u << 25);     // pad_amount = 4 DWORDs (8 bf16 -> 40-elem stride)
    g1[1] = (td0 & 0xFFFFu) << 16;                     // tensor_dim0[15:0]
    g1[2] = (td0 >> 16) | ((td1 & 0xFFFFu) << 16);     // tensor_dim0[31:16] | tensor_dim1[15:0]
    g1[3] = (td1 >> 16) | (32u << 16);                 // tensor_dim1[31:16] | tile_dim0=32
    g1[4] = rows;                                      // tile_dim1 (tile_dim2=0)
    g1[5] = stride_elems;                              // tensor_dim0_stride[31:0]
    g1[6] = 0u;                                        // stride[47:32] | dim1_stride lo
    g1[7] = 0u;
    asm volatile("tensor_load_to_lds %0, %1" :: "s"(g0), "s"(g1) : "memory");
}

// ---------------------------------------------------------------------------
// Weight convert+transpose: in fp32 [K,N] row-major -> out bf16 [N,K] row-major
// ---------------------------------------------------------------------------
__global__ void wtrans_kernel(const float* __restrict__ in, bf16* __restrict__ out,
                              int K, int N) {
    __shared__ float tile[32][33];
    int bn = blockIdx.x * 32;
    int bk = blockIdx.y * 32;
    int tx = threadIdx.x & 31, ty = threadIdx.x >> 5;
    #pragma unroll
    for (int i = 0; i < 32; i += 8)
        tile[ty + i][tx] = in[(size_t)(bk + ty + i) * N + bn + tx];
    __syncthreads();
    #pragma unroll
    for (int i = 0; i < 32; i += 8)
        out[(size_t)(bn + ty + i) * K + bk + tx] = (bf16)tile[tx][ty + i];
}

// ---------------------------------------------------------------------------
// LayerNorm: fp32 [rows, D] -> bf16 [rows, D]; one block (256 thr) per row.
// ---------------------------------------------------------------------------
__global__ __launch_bounds__(256) void ln_kernel(const float* __restrict__ x,
                                                 const float* __restrict__ g,
                                                 const float* __restrict__ b,
                                                 bf16* __restrict__ out) {
    int row = blockIdx.x;
    int t = threadIdx.x;
    const float* xr = x + (size_t)row * D_;
    float v[4], s = 0.f, ss = 0.f;
    #pragma unroll
    for (int i = 0; i < 4; i++) { v[i] = xr[t + i * 256]; s += v[i]; ss += v[i] * v[i]; }
    #pragma unroll
    for (int m = 16; m >= 1; m >>= 1) { s += __shfl_xor(s, m, 32); ss += __shfl_xor(ss, m, 32); }
    __shared__ float red[2][8];
    int w = t >> 5;
    if ((t & 31) == 0) { red[0][w] = s; red[1][w] = ss; }
    __syncthreads();
    s = 0.f; ss = 0.f;
    #pragma unroll
    for (int i = 0; i < 8; i++) { s += red[0][i]; ss += red[1][i]; }
    float mu = s * (1.0f / D_);
    float var = ss * (1.0f / D_) - mu * mu;
    float rstd = rsqrtf(var + 1e-6f);
    bf16* orow = out + (size_t)row * D_;
    #pragma unroll
    for (int i = 0; i < 4; i++) {
        int c = t + i * 256;
        orow[c] = (bf16)((v[i] - mu) * rstd * g[c] + b[c]);
    }
}

// ---------------------------------------------------------------------------
// GEMM: C[M,N] = A[M,K] * Bt[N,K]^T (+bias) (+epilogue). bf16 in, f32 accum.
//   EPI 0: bias -> bf16 out
//   EPI 1: bias + f32 residual -> f32 out
//   EPI 2: bias + exact GELU -> bf16 out
// 128x128 block tile, TK=32, 8 waves (2x4), wave tile 64x32 (4x2 WMMA tiles).
// Tiles are DMA'd into double-buffered LDS by wave 0 via TDM; all waves only
// do ds_load fragments + v_wmma.
// ---------------------------------------------------------------------------
#define TM 128
#define TN 128
#define TK 32
#define LP 40   // padded LDS K-stride (elements) == TDM pad output stride

template <int EPI>
__global__ __launch_bounds__(256) void gemm_kernel(const bf16* __restrict__ A,
                                                   const bf16* __restrict__ Bt,
                                                   const float* __restrict__ bias,
                                                   const float* __restrict__ res,
                                                   void* __restrict__ Cout,
                                                   int M, int N, int K) {
    __shared__ bf16 As[2][TM][LP];
    __shared__ bf16 Bs[2][TN][LP];
    int tid = threadIdx.x;
    int lane = tid & 31, wave = tid >> 5;
    int l15 = lane & 15, lh = lane >> 4;
    int wm = wave >> 2, wn = wave & 3;                 // 2 x 4 wave grid
    int brow = blockIdx.y * TM, bcol = blockIdx.x * TN;

    f32x8 zero = {0.f, 0.f, 0.f, 0.f, 0.f, 0.f, 0.f, 0.f};
    f32x8 acc[4][2];
    #pragma unroll
    for (int mi = 0; mi < 4; mi++)
        #pragma unroll
        for (int ni = 0; ni < 2; ni++) acc[mi][ni] = zero;

    uint64_t abase = (uint64_t)(uintptr_t)A  + (uint64_t)brow * K * 2u;
    uint64_t bbase = (uint64_t)(uintptr_t)Bt + (uint64_t)bcol * K * 2u;
    uint32_t ldsA0 = (uint32_t)(uintptr_t)&As[0][0][0];
    uint32_t ldsA1 = (uint32_t)(uintptr_t)&As[1][0][0];
    uint32_t ldsB0 = (uint32_t)(uintptr_t)&Bs[0][0][0];
    uint32_t ldsB1 = (uint32_t)(uintptr_t)&Bs[1][0][0];

    int nk = K / TK;
    if (wave == 0) {                     // prefetch tile 0 into buffer 0
        tdm_load_tile32(ldsA0, abase, TM, (uint32_t)K);
        tdm_load_tile32(ldsB0, bbase, TN, (uint32_t)K);
    }

    for (int i = 0; i < nk; i++) {
        int p = i & 1;
        __syncthreads();                 // buffer p^1 free (iter i-1 readers done)
        if (wave == 0) {
            if (i + 1 < nk) {            // DMA next tile while we compute this one
                uint64_t off = (uint64_t)(i + 1) * TK * 2u;
                tdm_load_tile32(p ? ldsA0 : ldsA1, abase + off, TM, (uint32_t)K);
                tdm_load_tile32(p ? ldsB0 : ldsB1, bbase + off, TN, (uint32_t)K);
                __builtin_amdgcn_s_wait_tensorcnt(2);   // pair(i) done; pair(i+1) may fly
            } else {
                __builtin_amdgcn_s_wait_tensorcnt(0);
            }
        }
        __syncthreads();                 // publish buffer p

        FragU af[4], bfr[2];
        #pragma unroll
        for (int mi = 0; mi < 4; mi++) {
            const bf16* ptr = &As[p][wm * 64 + mi * 16 + l15][lh * 8];
            af[mi].u[0] = *(const uint4*)ptr;
            af[mi].u[1] = *(const uint4*)(ptr + 16);
        }
        #pragma unroll
        for (int ni = 0; ni < 2; ni++) {
            const bf16* ptr = &Bs[p][wn * 32 + ni * 16 + l15][lh * 8];
            bfr[ni].u[0] = *(const uint4*)ptr;
            bfr[ni].u[1] = *(const uint4*)(ptr + 16);
        }
        #pragma unroll
        for (int mi = 0; mi < 4; mi++)
            #pragma unroll
            for (int ni = 0; ni < 2; ni++)
                acc[mi][ni] = wmma_bf16(af[mi], bfr[ni], acc[mi][ni]);
    }

    // ---- epilogue ----
    #pragma unroll
    for (int mi = 0; mi < 4; mi++) {
        #pragma unroll
        for (int ni = 0; ni < 2; ni++) {
            int col = bcol + wn * 32 + ni * 16 + l15;
            float bv = bias[col];
            #pragma unroll
            for (int r = 0; r < 8; r++) {
                int row = brow + wm * 64 + mi * 16 + lh * 8 + r;
                size_t idx = (size_t)row * N + col;
                float c = acc[mi][ni][r] + bv;
                if (EPI == 0) {
                    ((bf16*)Cout)[idx] = (bf16)c;
                } else if (EPI == 1) {
                    ((float*)Cout)[idx] = c + res[idx];
                } else {
                    float ge = 0.5f * c * (1.0f + erff(c * 0.70710678118f));
                    ((bf16*)Cout)[idx] = (bf16)ge;
                }
            }
        }
    }
}

// ---------------------------------------------------------------------------
// L2-normalize per-head rows of q and k in place (bf16, row = 64 elems).
// ---------------------------------------------------------------------------
__global__ void l2norm_kernel(bf16* __restrict__ q, bf16* __restrict__ k) {
    int idx = blockIdx.x * blockDim.x + threadIdx.x;   // TOK*H*2 threads
    bf16* p = ((idx & 1) ? k : q) + (size_t)(idx >> 1) * HD_;
    uint4 u[8];
    uint4* pu = (uint4*)p;
    #pragma unroll
    for (int i = 0; i < 8; i++) u[i] = pu[i];
    bf16* e = (bf16*)u;
    float ss = 0.f;
    #pragma unroll
    for (int i = 0; i < 64; i++) { float v = (float)e[i]; ss += v * v; }
    float inv = 1.0f / fmaxf(sqrtf(ss), 1e-12f);
    #pragma unroll
    for (int i = 0; i < 64; i++) e[i] = (bf16)((float)e[i] * inv);
    #pragma unroll
    for (int i = 0; i < 8; i++) pu[i] = u[i];
}

// ---------------------------------------------------------------------------
// Per-head transpose of V: v[b, s, h*64 + d] -> vt[((b*H+h)*64 + d)*S + s]
// ---------------------------------------------------------------------------
__global__ __launch_bounds__(256) void vtrans_kernel(const bf16* __restrict__ v,
                                                     bf16* __restrict__ vt) {
    __shared__ bf16 tile[64][72];
    int bh = blockIdx.y, b = bh >> 4, h = bh & 15;
    int s0 = blockIdx.x * 64;
    int tid = threadIdx.x;
    int row = tid >> 2, seg = (tid & 3) * 16;
    const bf16* src = v + (size_t)(b * S_ + s0 + row) * D_ + h * HD_ + seg;
    *(uint4*)&tile[row][seg]     = *(const uint4*)src;
    *(uint4*)&tile[row][seg + 8] = *(const uint4*)(src + 8);
    __syncthreads();
    int d = tid >> 2, sseg = (tid & 3) * 16;
    bf16 tmp[16];
    #pragma unroll
    for (int i = 0; i < 16; i++) tmp[i] = tile[sseg + i][d];
    bf16* dst = vt + (size_t)((b * H_ + h) * HD_ + d) * S_ + s0 + sseg;
    *(uint4*)dst       = *(uint4*)tmp;
    *(uint4*)(dst + 8) = *(uint4*)(tmp + 8);
}

// ---------------------------------------------------------------------------
// Streaming cosine attention. Scores are bounded by +scale, so softmax uses a
// fixed max c = scale (exact): p = exp(scale*(dot-1)); out = sum(p*V)/sum(p).
// Grid (S/64, B*H), 128 threads = 4 waves; wave owns 16 query rows.
// ---------------------------------------------------------------------------
__global__ __launch_bounds__(128) void attn_kernel(const bf16* __restrict__ q,
                                                   const bf16* __restrict__ k,
                                                   const bf16* __restrict__ vt,
                                                   const float* __restrict__ logit_scale,
                                                   bf16* __restrict__ ctx) {
    __shared__ bf16 Qs[64][72];
    __shared__ bf16 Ks[64][72];
    __shared__ bf16 Vts[64][72];
    __shared__ bf16 Ps[4][16][72];

    int bh = blockIdx.y, b = bh >> 4, h = bh & 15;
    int q0 = blockIdx.x * 64;
    int tid = threadIdx.x, lane = tid & 31, wave = tid >> 5;
    int l15 = lane & 15, lh = lane >> 4;
    float scale = __expf(fminf(logit_scale[h], LOG_MAX_));

    {   // load Q tile (64 x 64)
        int row = tid >> 1, seg = (tid & 1) * 32;
        const bf16* src = q + (size_t)(b * S_ + q0 + row) * D_ + h * HD_ + seg;
        const uint4* s4 = (const uint4*)src;
        uint4 u0 = s4[0], u1 = s4[1], u2 = s4[2], u3 = s4[3];
        *(uint4*)&Qs[row][seg]      = u0;
        *(uint4*)&Qs[row][seg + 8]  = u1;
        *(uint4*)&Qs[row][seg + 16] = u2;
        *(uint4*)&Qs[row][seg + 24] = u3;
    }
    __syncthreads();

    FragU qa[2];
    #pragma unroll
    for (int kk = 0; kk < 2; kk++) {
        const bf16* p = &Qs[wave * 16 + l15][kk * 32 + lh * 8];
        qa[kk].u[0] = *(const uint4*)p;
        qa[kk].u[1] = *(const uint4*)(p + 16);
    }

    f32x8 zero = {0.f, 0.f, 0.f, 0.f, 0.f, 0.f, 0.f, 0.f};
    f32x8 oacc[4];
    #pragma unroll
    for (int ni = 0; ni < 4; ni++) oacc[ni] = zero;
    float lsum[8] = {0.f, 0.f, 0.f, 0.f, 0.f, 0.f, 0.f, 0.f};

    int ldrow = tid >> 1, ldseg = (tid & 1) * 32;

    for (int kb = 0; kb < S_ / 64; kb++) {
        __syncthreads();
        {   // load K tile [key][d] and Vt tile [d][key]
            const bf16* ks = k + (size_t)(b * S_ + kb * 64 + ldrow) * D_ + h * HD_ + ldseg;
            const uint4* k4 = (const uint4*)ks;
            uint4 u0 = k4[0], u1 = k4[1], u2 = k4[2], u3 = k4[3];
            *(uint4*)&Ks[ldrow][ldseg]      = u0;
            *(uint4*)&Ks[ldrow][ldseg + 8]  = u1;
            *(uint4*)&Ks[ldrow][ldseg + 16] = u2;
            *(uint4*)&Ks[ldrow][ldseg + 24] = u3;
            const bf16* vs = vt + (size_t)((b * H_ + h) * HD_ + ldrow) * S_ + kb * 64 + ldseg;
            const uint4* v4 = (const uint4*)vs;
            uint4 w0 = v4[0], w1 = v4[1], w2 = v4[2], w3 = v4[3];
            *(uint4*)&Vts[ldrow][ldseg]      = w0;
            *(uint4*)&Vts[ldrow][ldseg + 8]  = w1;
            *(uint4*)&Vts[ldrow][ldseg + 16] = w2;
            *(uint4*)&Vts[ldrow][ldseg + 24] = w3;
        }
        __syncthreads();

        f32x8 sacc[4];
        #pragma unroll
        for (int ni = 0; ni < 4; ni++) sacc[ni] = zero;
        #pragma unroll
        for (int kk = 0; kk < 2; kk++) {
            #pragma unroll
            for (int ni = 0; ni < 4; ni++) {
                FragU bb;
                const bf16* p = &Ks[ni * 16 + l15][kk * 32 + lh * 8];
                bb.u[0] = *(const uint4*)p;
                bb.u[1] = *(const uint4*)(p + 16);
                sacc[ni] = wmma_bf16(qa[kk], bb, sacc[ni]);
            }
        }

        #pragma unroll
        for (int ni = 0; ni < 4; ni++)
            #pragma unroll
            for (int r = 0; r < 8; r++)
                sacc[ni][r] = __expf(scale * (sacc[ni][r] - 1.0f));
        #pragma unroll
        for (int r = 0; r < 8; r++) {
            float s = sacc[0][r] + sacc[1][r] + sacc[2][r] + sacc[3][r];
            s += __shfl_xor(s, 1, 32);
            s += __shfl_xor(s, 2, 32);
            s += __shfl_xor(s, 4, 32);
            s += __shfl_xor(s, 8, 32);
            lsum[r] += s;
        }
        #pragma unroll
        for (int ni = 0; ni < 4; ni++)
            #pragma unroll
            for (int r = 0; r < 8; r++)
                Ps[wave][r + lh * 8][ni * 16 + l15] = (bf16)sacc[ni][r];
        __syncthreads();

        #pragma unroll
        for (int kk = 0; kk < 2; kk++) {
            FragU pa;
            const bf16* pp = &Ps[wave][l15][kk * 32 + lh * 8];
            pa.u[0] = *(const uint4*)pp;
            pa.u[1] = *(const uint4*)(pp + 16);
            #pragma unroll
            for (int ni = 0; ni < 4; ni++) {
                FragU vb;
                const bf16* vp = &Vts[ni * 16 + l15][kk * 32 + lh * 8];
                vb.u[0] = *(const uint4*)vp;
                vb.u[1] = *(const uint4*)(vp + 16);
                oacc[ni] = wmma_bf16(pa, vb, oacc[ni]);
            }
        }
    }

    #pragma unroll
    for (int ni = 0; ni < 4; ni++) {
        #pragma unroll
        for (int r = 0; r < 8; r++) {
            int row = q0 + wave * 16 + lh * 8 + r;
            int d = ni * 16 + l15;
            float o = oacc[ni][r] / lsum[r];
            ctx[(size_t)(b * S_ + row) * D_ + h * HD_ + d] = (bf16)o;
        }
    }
}

// ---------------------------------------------------------------------------
// Launch
// ---------------------------------------------------------------------------
extern "C" void kernel_launch(void* const* d_in, const int* in_sizes, int n_in,
                              void* d_out, int out_size, void* d_ws, size_t ws_size,
                              hipStream_t stream) {
    (void)in_sizes; (void)n_in; (void)out_size; (void)ws_size;
    const float* x    = (const float*)d_in[0];
    const float* wq   = (const float*)d_in[1];
    const float* bq   = (const float*)d_in[2];
    const float* wk   = (const float*)d_in[3];
    const float* bk   = (const float*)d_in[4];
    const float* wv   = (const float*)d_in[5];
    const float* bv   = (const float*)d_in[6];
    const float* wo   = (const float*)d_in[7];
    const float* bo   = (const float*)d_in[8];
    const float* w1   = (const float*)d_in[9];
    const float* b1   = (const float*)d_in[10];
    const float* w2   = (const float*)d_in[11];
    const float* b2   = (const float*)d_in[12];
    const float* ln1g = (const float*)d_in[13];
    const float* ln1b = (const float*)d_in[14];
    const float* ln2g = (const float*)d_in[15];
    const float* ln2b = (const float*)d_in[16];
    const float* lsc  = (const float*)d_in[17];
    float* out = (float*)d_out;

    char* w = (char*)d_ws;
    auto alloc = [&](size_t bytes) -> char* {
        char* p = w;
        w += (bytes + 255) & ~(size_t)255;
        return p;
    };
    bf16*  wqt  = (bf16*)alloc((size_t)D_ * D_ * 2);
    bf16*  wkt  = (bf16*)alloc((size_t)D_ * D_ * 2);
    bf16*  wvt  = (bf16*)alloc((size_t)D_ * D_ * 2);
    bf16*  wot  = (bf16*)alloc((size_t)D_ * D_ * 2);
    bf16*  w1t  = (bf16*)alloc((size_t)D_ * MLP_ * 2);
    bf16*  w2t  = (bf16*)alloc((size_t)MLP_ * D_ * 2);
    bf16*  xn   = (bf16*)alloc((size_t)TOK * D_ * 2);
    bf16*  qb   = (bf16*)alloc((size_t)TOK * D_ * 2);
    bf16*  kb   = (bf16*)alloc((size_t)TOK * D_ * 2);
    bf16*  vb   = (bf16*)alloc((size_t)TOK * D_ * 2);
    bf16*  vtb  = (bf16*)alloc((size_t)TOK * D_ * 2);
    bf16*  ctxb = (bf16*)alloc((size_t)TOK * D_ * 2);
    float* aout = (float*)alloc((size_t)TOK * D_ * 4);
    bf16*  xn2  = (bf16*)alloc((size_t)TOK * D_ * 2);
    bf16*  h1   = (bf16*)alloc((size_t)TOK * MLP_ * 2);

    wtrans_kernel<<<dim3(D_ / 32, D_ / 32), 256, 0, stream>>>(wq, wqt, D_, D_);
    wtrans_kernel<<<dim3(D_ / 32, D_ / 32), 256, 0, stream>>>(wk, wkt, D_, D_);
    wtrans_kernel<<<dim3(D_ / 32, D_ / 32), 256, 0, stream>>>(wv, wvt, D_, D_);
    wtrans_kernel<<<dim3(D_ / 32, D_ / 32), 256, 0, stream>>>(wo, wot, D_, D_);
    wtrans_kernel<<<dim3(MLP_ / 32, D_ / 32), 256, 0, stream>>>(w1, w1t, D_, MLP_);
    wtrans_kernel<<<dim3(D_ / 32, MLP_ / 32), 256, 0, stream>>>(w2, w2t, MLP_, D_);

    ln_kernel<<<TOK, 256, 0, stream>>>(x, ln1g, ln1b, xn);

    gemm_kernel<0><<<dim3(D_ / TN, TOK / TM), 256, 0, stream>>>(xn, wqt, bq, nullptr, qb, TOK, D_, D_);
    gemm_kernel<0><<<dim3(D_ / TN, TOK / TM), 256, 0, stream>>>(xn, wkt, bk, nullptr, kb, TOK, D_, D_);
    gemm_kernel<0><<<dim3(D_ / TN, TOK / TM), 256, 0, stream>>>(xn, wvt, bv, nullptr, vb, TOK, D_, D_);

    l2norm_kernel<<<(TOK * H_ * 2) / 256, 256, 0, stream>>>(qb, kb);
    vtrans_kernel<<<dim3(S_ / 64, B_ * H_), 256, 0, stream>>>(vb, vtb);

    attn_kernel<<<dim3(S_ / 64, B_ * H_), 128, 0, stream>>>(qb, kb, vtb, lsc, ctxb);

    gemm_kernel<1><<<dim3(D_ / TN, TOK / TM), 256, 0, stream>>>(ctxb, wot, bo, x, aout, TOK, D_, D_);

    ln_kernel<<<TOK, 256, 0, stream>>>(aout, ln2g, ln2b, xn2);

    gemm_kernel<2><<<dim3(MLP_ / TN, TOK / TM), 256, 0, stream>>>(xn2, w1t, b1, nullptr, h1, TOK, MLP_, D_);
    gemm_kernel<1><<<dim3(D_ / TN, TOK / TM), 256, 0, stream>>>(h1, w2t, b2, aout, out, TOK, D_, MLP_);
}
